// CausalSelfAttention_30571577213406
// MI455X (gfx1250) — matrix operand order repaired and checked by reading, not compiled, not run on hardware
//
#include <hip/hip_runtime.h>
#include <hip/hip_bf16.h>

typedef __attribute__((ext_vector_type(16))) _Float16 v16h;
typedef __attribute__((ext_vector_type(8)))  float    v8f;
typedef __attribute__((ext_vector_type(4)))  unsigned u32x4;
typedef __attribute__((ext_vector_type(8)))  int      i32x8;
typedef __attribute__((ext_vector_type(4)))  int      i32x4;

#define D_MODEL 1024
#define N_HEADS 16
#define D_HEAD  64
#define SEQ     2048
#define BATCH   4
#define ROWS    (BATCH * SEQ)   // 8192
#define QKV_N   (3 * D_MODEL)   // 3072

#define WMMA_F16(a, b, c) \
  __builtin_amdgcn_wmma_f32_16x16x32_f16(false, (a), false, (b), (short)0, (c), false, false)

// ---------------- f32 -> f16 conversion ----------------
__global__ void cvt_f32_to_f16(const float* __restrict__ in, _Float16* __restrict__ out, int n) {
  int i = blockIdx.x * blockDim.x + threadIdx.x;
  int stride = gridDim.x * blockDim.x;
  for (; i < n; i += stride) out[i] = (_Float16)in[i];
}

// ---------------- WMMA fragment loaders (wave32 layouts, CDNA5 ISA) ----
__device__ __forceinline__ int a_koff(int v, int h) {
  return 16 * h + 2 * v;
}

// A fragment from row-major [16 x >=32] tile at p, leading dim ld (halves)
__device__ __forceinline__ v16h load_frag_A_rowmajor(const _Float16* p, int ld) {
  int lane = threadIdx.x & 31;
  int row = lane & 15, h = lane >> 4;
  v16h a;
#pragma unroll
  for (int v = 0; v < 8; ++v) {
    int k = a_koff(v, h);
    a[2 * v]     = p[(size_t)row * ld + k];
    a[2 * v + 1] = p[(size_t)row * ld + k + 1];
  }
  return a;
}

// B fragment (B[k][n], 32x16) where memory is row-major [n][k] (i.e. B = Mem^T)
__device__ __forceinline__ v16h load_frag_B_transposed(const _Float16* p, int ld) {
  int lane = threadIdx.x & 31;
  int col = lane & 15, h = lane >> 4;
  v16h b;
#pragma unroll
  for (int v = 0; v < 8; ++v) {
    int k = 16 * h + 2 * v;
    b[2 * v]     = p[(size_t)col * ld + k];
    b[2 * v + 1] = p[(size_t)col * ld + k + 1];
  }
  return b;
}

// B fragment (B[k][n], 32x16) where memory is row-major [k][n]
__device__ __forceinline__ v16h load_frag_B_rowmajor(const _Float16* p, int ld) {
  int lane = threadIdx.x & 31;
  int col = lane & 15, h = lane >> 4;
  v16h b;
#pragma unroll
  for (int v = 0; v < 8; ++v) {
    int k = 16 * h + 2 * v;
    b[2 * v]     = p[(size_t)k * ld + col];
    b[2 * v + 1] = p[(size_t)(k + 1) * ld + col];
  }
  return b;
}

// ---------------- Tensor Data Mover: load a [tile_rows x 32] f16 tile to LDS ----
// D# per CDNA5 ISA. LDS rows padded: 16 dwords data + 4 pad
// (pad_interval=3 -> every 16 dwords, pad_amount=3 -> 4 dwords) => 40-half stride.
__device__ __forceinline__ void tdm_load_tile_f16(const _Float16* gsrc, unsigned lds_byte_addr,
                                                  unsigned rows_total, unsigned k_total,
                                                  unsigned tile_rows) {
  unsigned long long ga = (unsigned long long)(size_t)gsrc;
  u32x4 g0;
  g0[0] = 1u;                                          // count=1 (valid), user mode
  g0[1] = lds_byte_addr;                               // LDS dest (bytes)
  g0[2] = (unsigned)(ga & 0xFFFFFFFFu);                // global_addr[31:0]
  g0[3] = (unsigned)((ga >> 32) & 0x1FFFFFFu) | (2u << 30); // addr[56:32] | type=2
  i32x8 g1;
  g1[0] = (int)((1u << 16)      // data_size = 1 -> 2 bytes
              | (1u << 20)      // pad_enable
              | (3u << 22)      // pad_interval: 16 dwords
              | (3u << 25));    // pad_amount: 4 dwords
  g1[1] = (int)((k_total & 0xFFFFu) << 16);                          // tensor_dim0 lo
  g1[2] = (int)(((k_total >> 16) & 0xFFFFu) | ((rows_total & 0xFFFFu) << 16)); // dim0 hi | dim1 lo
  g1[3] = (int)(((rows_total >> 16) & 0xFFFFu) | (32u << 16));       // dim1 hi | tile_dim0=32
  g1[4] = (int)(tile_rows & 0xFFFFu);                                // tile_dim1 (tile_dim2=0)
  g1[5] = (int)k_total;                                              // tensor_dim0_stride lo32
  g1[6] = 0;
  g1[7] = 0;
  i32x4 z4 = {0, 0, 0, 0};
  i32x8 z8 = {0, 0, 0, 0, 0, 0, 0, 0};
  __builtin_amdgcn_tensor_load_to_lds(g0, g1, z4, z4, z8, 0);
}

__device__ __forceinline__ unsigned lds_offset_of(const void* p) {
  // generic pointer low 32 bits == LDS byte address (flat->LDS mapping uses addr[31:0])
  return (unsigned)(size_t)p;
}

// ---------------- GEMM: C[M,N] = A[M,K] (f16) @ W[N,K]^T (f16) ----------------
// 256 threads = 8 waves; 128x128 block tile; wave tile 32x64 (8 WMMA / K-step).
// Double-buffered TDM pipeline: wave 0 issues next tile's DMA before computing the
// current tile, waits TENSORcnt only after compute -> DMA overlaps the WMMAs.
template <bool OUT_HALF>
__global__ __launch_bounds__(256) void gemm_xwT(const _Float16* __restrict__ A,
                                                const _Float16* __restrict__ W,
                                                _Float16* __restrict__ Ch,
                                                float* __restrict__ Cf,
                                                int M, int N, int K) {
  const int LDT = 40; // halves: 32 data + 8 pad per row (TDM pad writes this layout)
  __shared__ __align__(16) _Float16 As[2][128 * LDT];
  __shared__ __align__(16) _Float16 Ws[2][128 * LDT];

  int n0 = blockIdx.x * 128;
  int m0 = blockIdx.y * 128;
  int tid = threadIdx.x;
  int wid = tid >> 5;
  int wm = (wid >> 1) * 32;   // 0,32,64,96
  int wn = (wid & 1) * 64;    // 0,64

  unsigned as_lds[2] = { lds_offset_of(As[0]), lds_offset_of(As[1]) };
  unsigned ws_lds[2] = { lds_offset_of(Ws[0]), lds_offset_of(Ws[1]) };

  const _Float16* Arow = A + (size_t)m0 * K;
  const _Float16* Wrow = W + (size_t)n0 * K;

  // prologue: fill buffer 0
  if (wid == 0) {
    tdm_load_tile_f16(Arow, as_lds[0], (unsigned)M, (unsigned)K, 128u);
    tdm_load_tile_f16(Wrow, ws_lds[0], (unsigned)N, (unsigned)K, 128u);
    __builtin_amdgcn_s_wait_tensorcnt(0);
  }
  __syncthreads();

  v8f acc[2][4] = {};
  int cur = 0;

  for (int k0 = 0; k0 < K; k0 += 32) {
    int nxt = cur ^ 1;
    // kick off DMA for the next K-slab while we compute this one
    if (wid == 0 && k0 + 32 < K) {
      tdm_load_tile_f16(Arow + k0 + 32, as_lds[nxt], (unsigned)M, (unsigned)K, 128u);
      tdm_load_tile_f16(Wrow + k0 + 32, ws_lds[nxt], (unsigned)N, (unsigned)K, 128u);
    }

    v16h a0 = load_frag_A_rowmajor(&As[cur][(size_t)(wm + 0) * LDT], LDT);
    v16h a1 = load_frag_A_rowmajor(&As[cur][(size_t)(wm + 16) * LDT], LDT);
#pragma unroll
    for (int nt = 0; nt < 4; ++nt) {
      v16h b = load_frag_B_transposed(&Ws[cur][(size_t)(wn + nt * 16) * LDT], LDT);
      acc[0][nt] = WMMA_F16(a0, b, acc[0][nt]);
      acc[1][nt] = WMMA_F16(a1, b, acc[1][nt]);
    }

    if (wid == 0) __builtin_amdgcn_s_wait_tensorcnt(0); // next buffer resident
    __syncthreads(); // also fences this buffer's LDS reads before it is reused
    cur = nxt;
  }

  // Epilogue: C/D layout: VGPR r -> row r + 8*(lane>=16); col = lane&15
  int lane = tid & 31, h = lane >> 4, col = lane & 15;
#pragma unroll
  for (int mt = 0; mt < 2; ++mt)
#pragma unroll
    for (int nt = 0; nt < 4; ++nt)
#pragma unroll
      for (int r = 0; r < 8; ++r) {
        size_t row = (size_t)m0 + wm + mt * 16 + r + 8 * h;
        size_t cc  = (size_t)n0 + wn + nt * 16 + col;
        if (OUT_HALF) Ch[row * N + cc] = (_Float16)acc[mt][nt][r];
        else          Cf[row * N + cc] = acc[mt][nt][r];
      }
}

// ---------------- Flash attention: one wave per (b, h, 16-query tile) ----------
__global__ __launch_bounds__(32) void attn_fa(const _Float16* __restrict__ qkv,
                                              _Float16* __restrict__ aout) {
  int qt = blockIdx.x;
  int hd = blockIdx.y;
  int b  = blockIdx.z;
  int q0 = qt * 16;
  const int ld = QKV_N;

  const _Float16* Qb = qkv + (size_t)b * SEQ * QKV_N + (size_t)hd * D_HEAD;
  const _Float16* Kb = Qb + D_MODEL;
  const _Float16* Vb = Qb + 2 * D_MODEL;

  int lane = threadIdx.x & 31;
  int h = lane >> 4, col = lane & 15;

  v16h qa0 = load_frag_A_rowmajor(Qb + (size_t)q0 * ld, ld);
  v16h qa1 = load_frag_A_rowmajor(Qb + (size_t)q0 * ld + 32, ld);

  v8f o[4] = {};
  float m_i[8], l_i[8];
#pragma unroll
  for (int r = 0; r < 8; ++r) { m_i[r] = -__builtin_inff(); l_i[r] = 0.f; }

  __shared__ __align__(16) _Float16 Psm[16 * 32];

  const float scale = 0.125f; // 1/sqrt(64)
  const int jend = q0 + 16;

  for (int j = 0; j < jend; j += 32) {
    // prefetch next K/V tiles into cache while we do WMMA + softmax on this one
    if (j + 32 < jend) {
      __builtin_prefetch((const void*)(Kb + (size_t)(j + 32) * ld), 0, 3);
      __builtin_prefetch((const void*)(Vb + (size_t)(j + 32) * ld), 0, 3);
    }

    // ---- S[16x32] = Q @ K^T over Dh=64
    v8f s0 = {}, s1 = {};
    {
      v16h kb = load_frag_B_transposed(Kb + (size_t)j * ld, ld);
      s0 = WMMA_F16(qa0, kb, s0);
      kb = load_frag_B_transposed(Kb + (size_t)j * ld + 32, ld);
      s0 = WMMA_F16(qa1, kb, s0);
      kb = load_frag_B_transposed(Kb + (size_t)(j + 16) * ld, ld);
      s1 = WMMA_F16(qa0, kb, s1);
      kb = load_frag_B_transposed(Kb + (size_t)(j + 16) * ld + 32, ld);
      s1 = WMMA_F16(qa1, kb, s1);
    }

    // ---- causal mask + online softmax (row reductions across 16 lanes)
    float p0[8], p1[8], rsc[8];
#pragma unroll
    for (int r = 0; r < 8; ++r) {
      int row = q0 + r + 8 * h;
      float v0 = (j + col > row)      ? -__builtin_inff() : s0[r] * scale;
      float v1 = (j + 16 + col > row) ? -__builtin_inff() : s1[r] * scale;
      float mx = fmaxf(v0, v1);
      mx = fmaxf(mx, __shfl_xor(mx, 1, 32));
      mx = fmaxf(mx, __shfl_xor(mx, 2, 32));
      mx = fmaxf(mx, __shfl_xor(mx, 4, 32));
      mx = fmaxf(mx, __shfl_xor(mx, 8, 32));
      float mn = fmaxf(m_i[r], mx);
      float e0 = __expf(v0 - mn), e1 = __expf(v1 - mn);
      float sum = e0 + e1;
      sum += __shfl_xor(sum, 1, 32);
      sum += __shfl_xor(sum, 2, 32);
      sum += __shfl_xor(sum, 4, 32);
      sum += __shfl_xor(sum, 8, 32);
      float rf = __expf(m_i[r] - mn);
      l_i[r] = l_i[r] * rf + sum;
      m_i[r] = mn;
      p0[r] = e0; p1[r] = e1; rsc[r] = rf;
    }

#pragma unroll
    for (int dt = 0; dt < 4; ++dt)
#pragma unroll
      for (int r = 0; r < 8; ++r) o[dt][r] *= rsc[r];

    // ---- P: C-layout -> LDS -> A-fragment layout
#pragma unroll
    for (int r = 0; r < 8; ++r) {
      int row = r + 8 * h;
      Psm[row * 32 + col]      = (_Float16)p0[r];
      Psm[row * 32 + col + 16] = (_Float16)p1[r];
    }
    __syncthreads();
    v16h pa = load_frag_A_rowmajor(Psm, 32);
    __syncthreads();

    // ---- O[16x64] += P[16x32] @ V[32x64]
#pragma unroll
    for (int dt = 0; dt < 4; ++dt) {
      v16h vb = load_frag_B_rowmajor(Vb + (size_t)j * ld + dt * 16, ld);
      o[dt] = WMMA_F16(pa, vb, o[dt]);
    }
  }

  _Float16* Ob = aout + (size_t)b * SEQ * D_MODEL + (size_t)hd * D_HEAD;
#pragma unroll
  for (int r = 0; r < 8; ++r) {
    float inv = 1.0f / l_i[r];
    int row = q0 + r + 8 * h;
#pragma unroll
    for (int dt = 0; dt < 4; ++dt)
      Ob[(size_t)row * D_MODEL + dt * 16 + col] = (_Float16)(o[dt][r] * inv);
  }
}

// ---------------- host-side launch ----------------
extern "C" void kernel_launch(void* const* d_in, const int* in_sizes, int n_in,
                              void* d_out, int out_size, void* d_ws, size_t ws_size,
                              hipStream_t stream) {
  (void)in_sizes; (void)n_in; (void)out_size; (void)ws_size;
  const float* x      = (const float*)d_in[0];
  const float* w_qkv  = (const float*)d_in[1];
  const float* w_proj = (const float*)d_in[2];
  float* out = (float*)d_out;

  char* ws = (char*)d_ws;
  _Float16* xh   = (_Float16*)(ws + 0);           // 8192*1024 f16 = 16 MB
  _Float16* wqh  = (_Float16*)(ws + 16777216);    // 3072*1024 f16 =  6 MB
  _Float16* wph  = (_Float16*)(ws + 23068672);    // 1024*1024 f16 =  2 MB
  _Float16* qkvh = (_Float16*)(ws + 25165824);    // 8192*3072 f16 = 48 MB
  _Float16* aoh  = (_Float16*)(ws + 75497472);    // 8192*1024 f16 = 16 MB

  cvt_f32_to_f16<<<1024, 256, 0, stream>>>(x, xh, ROWS * D_MODEL);
  cvt_f32_to_f16<<<1024, 256, 0, stream>>>(w_qkv, wqh, QKV_N * D_MODEL);
  cvt_f32_to_f16<<<512, 256, 0, stream>>>(w_proj, wph, D_MODEL * D_MODEL);

  gemm_xwT<true><<<dim3(QKV_N / 128, ROWS / 128), 256, 0, stream>>>(
      xh, wqh, qkvh, nullptr, ROWS, QKV_N, D_MODEL);

  attn_fa<<<dim3(SEQ / 16, N_HEADS, BATCH), 32, 0, stream>>>(qkvh, aoh);

  gemm_xwT<false><<<dim3(D_MODEL / 128, ROWS / 128), 256, 0, stream>>>(
      aoh, wph, nullptr, out, ROWS, D_MODEL, D_MODEL);
}
